// GINBackbone_33921651703941
// MI455X (gfx1250) — compile-verified, matrix operand order checked by reading, not run on hardware
//
#include <hip/hip_runtime.h>

#define EMB 300
#define HID 600
#define NLAYER 5
#define BN_EPS 1e-5f
#define KP1 320   // EMB padded to 32
#define KP2 608   // HID padded to 32
#define NCP1 640  // HID cols padded to 128
#define NCP2 384  // EMB cols padded to 128

typedef __attribute__((ext_vector_type(16))) __bf16 v16bf;
typedef __attribute__((ext_vector_type(8)))  float  v8f;
typedef int v4i_gcc __attribute__((vector_size(16)));   // matches builtin param type

union FragBF { v16bf v; unsigned int u[8]; };

// ---- gfx1250 async global->LDS path (ASYNCcnt-tracked), with fallback ----
#if defined(__has_builtin)
#if __has_builtin(__builtin_amdgcn_global_load_async_to_lds_b128) && \
    __has_builtin(__builtin_amdgcn_s_wait_asynccnt)
#define USE_ASYNC_LDS 1
#endif
#endif
#ifndef USE_ASYNC_LDS
#define USE_ASYNC_LDS 0
#endif

__device__ __forceinline__ void copy16(unsigned short* lds, const unsigned short* g) {
#if USE_ASYNC_LDS
  __builtin_amdgcn_global_load_async_to_lds_b128(
      (__attribute__((address_space(1))) v4i_gcc*)g,
      (__attribute__((address_space(3))) v4i_gcc*)lds, 0, 0);
#else
  *(uint4*)lds = *(const uint4*)g;
#endif
}

__device__ __forceinline__ unsigned short f2bf_bits(float f) {
  unsigned int u = __float_as_uint(f);
  u += 0x7FFFu + ((u >> 16) & 1u);   // round-to-nearest-even
  return (unsigned short)(u >> 16);
}

// ISA 16-bit A-operand layout (16x32): lane half h, VGPR j holds K pair at:
//   j<4 : 2j + 8h ;  j>=4 : 16 + 2(j-4) + 8h
__device__ __forceinline__ int kpair(int j, int half) {
  return ((j < 4) ? (2 * j) : (16 + 2 * (j - 4))) + half * 8;
}

// ---------------------------------------------------------------- elementwise
__global__ void zero_kernel(float* __restrict__ p, int n) {
  int i = blockIdx.x * blockDim.x + threadIdx.x;
  if (i < n) p[i] = 0.0f;
}

__global__ void node_emb_kernel(const int* __restrict__ x,
                                const float* __restrict__ xe1,
                                const float* __restrict__ xe2,
                                float* __restrict__ h, int N) {
  int i = blockIdx.x * blockDim.x + threadIdx.x;
  if (i >= N * EMB) return;
  int n = i / EMB, c = i - n * EMB;
  h[i] = xe1[(size_t)x[2 * n] * EMB + c] + xe2[(size_t)x[2 * n + 1] * EMB + c];
}

// msg = h[src] + edge_emb1[bond_type] + edge_emb2[bond_dir]; atomic scatter to dst
__global__ void scatter_kernel(const float* __restrict__ h,
                               const int* __restrict__ ei,   // [2,E]
                               const int* __restrict__ ea,   // [E,2]
                               const float* __restrict__ e1, // [6,EMB]
                               const float* __restrict__ e2, // [3,EMB]
                               float* __restrict__ aggr, int N, int E) {
  int i = blockIdx.x * blockDim.x + threadIdx.x;
  int total = (E + N) * EMB;
  if (i >= total) return;
  int e = i / EMB, c = i - e * EMB;
  int s, d, bt, bd;
  if (e < E) { s = ei[e]; d = ei[E + e]; bt = ea[2 * e]; bd = ea[2 * e + 1]; }
  else       { s = d = e - E; bt = 4; bd = 0; }   // self-loop: attr [4,0]
  float msg = h[(size_t)s * EMB + c] + e1[bt * EMB + c] + e2[bd * EMB + c];
  atomicAdd(&aggr[(size_t)d * EMB + c], msg);
}

// W [K,Ncol] f32 -> Wp [NcolPad,Kpad] bf16 (n-major, K-contiguous, zero-padded)
__global__ void pack_w_kernel(const float* __restrict__ W,
                              unsigned short* __restrict__ Wp,
                              int K, int Ncol, int Kpad, int NcolPad) {
  int i = blockIdx.x * blockDim.x + threadIdx.x;
  if (i >= NcolPad * Kpad) return;
  int c = i / Kpad, k = i - c * Kpad;
  unsigned short bits = 0;
  if (c < Ncol && k < K) bits = f2bf_bits(W[(size_t)k * Ncol + c]);
  Wp[i] = bits;
}

// aggr [N,EMB] f32 -> Ap [Npad,KP1] bf16 zero-padded
__global__ void cvt_a_kernel(const float* __restrict__ A,
                             unsigned short* __restrict__ Ap,
                             int N, int Npad) {
  int i = blockIdx.x * blockDim.x + threadIdx.x;
  if (i >= Npad * KP1) return;
  int r = i / KP1, k = i - r * KP1;
  unsigned short bits = 0;
  if (r < N && k < EMB) bits = f2bf_bits(A[(size_t)r * EMB + k]);
  Ap[i] = bits;
}

// ---------------------------------------------------------------- WMMA GEMM
// C = act(A * B + bias). A [Mpad,Kpad] bf16 (zero-padded), Bp [NcolPad,Kpad]
// bf16 packed transposed. 256 thr = 8 waves; block tile 64x128; wave 16x64
// (4 x v_wmma_f32_16x16x32_bf16 per 32-K step). Guard-free, double-buffered
// async global->LDS pipeline: next tile in flight while current tile computes.
template <bool RELU_BF16_OUT>
__global__ __launch_bounds__(256) void gemm_wmma_kernel(
    const unsigned short* __restrict__ A,
    const unsigned short* __restrict__ Bp,
    const float* __restrict__ bias, void* __restrict__ Cout,
    int Kpad, int ldc, int NcolStore, int MStore) {
  __shared__ unsigned short ldsA[2][64 * 32];   // [buf][row][k]
  __shared__ unsigned short ldsB[2][128 * 32];  // [buf][col][k]

  const int tid  = threadIdx.x;
  const int lane = tid & 31;
  const int wave = tid >> 5;
  const int wrow = wave & 3;
  const int wcol = wave >> 2;
  const int half = lane >> 4;
  const int l16  = lane & 15;

  const int rowBase = blockIdx.x * 64;
  const int colBase = blockIdx.y * 128;
  const unsigned short* Abase = A  + (size_t)rowBase * Kpad;
  const unsigned short* Bbase = Bp + (size_t)colBase * Kpad;

  // per-thread 16B copy slots: 4 threads per 64B row segment
  const int cr   = tid >> 2;          // 0..63
  const int seg8 = (tid & 3) * 8;     // halfword offset 0/8/16/24

  v8f acc[4];
  #pragma unroll
  for (int t = 0; t < 4; ++t)
    #pragma unroll
    for (int r = 0; r < 8; ++r) acc[t][r] = 0.0f;

  auto issue = [&](int bi, int k0) {
    copy16(&ldsA[bi][cr * 32 + seg8],        &Abase[(size_t)cr * Kpad + k0 + seg8]);
    copy16(&ldsB[bi][cr * 32 + seg8],        &Bbase[(size_t)cr * Kpad + k0 + seg8]);
    copy16(&ldsB[bi][(cr + 64) * 32 + seg8], &Bbase[(size_t)(cr + 64) * Kpad + k0 + seg8]);
  };

  const int ksteps = Kpad >> 5;
  issue(0, 0);                               // prologue: fill buffer 0
  for (int ks = 0; ks < ksteps; ++ks) {
    const int buf = ks & 1;
    if (ks + 1 < ksteps) {
      issue(buf ^ 1, (ks + 1) << 5);         // next tile in flight
      if (ks + 2 < ksteps) {                 // L2 prefetch 2 steps ahead
        const int k2 = (ks + 2) << 5;
        __builtin_prefetch(&Abase[(size_t)cr * Kpad + k2 + seg8], 0, 1);
        __builtin_prefetch(&Bbase[(size_t)cr * Kpad + k2 + seg8], 0, 1);
        __builtin_prefetch(&Bbase[(size_t)(cr + 64) * Kpad + k2 + seg8], 0, 1);
      }
#if USE_ASYNC_LDS
      __builtin_amdgcn_s_wait_asynccnt(3);   // in-order: oldest 3 (this buf) done
#endif
    } else {
#if USE_ASYNC_LDS
      __builtin_amdgcn_s_wait_asynccnt(0);
#endif
    }
    __syncthreads();                         // buf's tile visible to all waves

    FragBF a;
    #pragma unroll
    for (int j = 0; j < 8; ++j)
      a.u[j] = *(const unsigned int*)
          &ldsA[buf][(wrow * 16 + l16) * 32 + kpair(j, half)];
    #pragma unroll
    for (int t = 0; t < 4; ++t) {
      FragBF b;
      const int cc = wcol * 64 + t * 16 + l16;
      #pragma unroll
      for (int j = 0; j < 8; ++j)
        b.u[j] = *(const unsigned int*)&ldsB[buf][cc * 32 + kpair(j, half)];
      acc[t] = __builtin_amdgcn_wmma_f32_16x16x32_bf16(
          false, a.v, false, b.v, (short)0, acc[t], false, false);
    }
    __syncthreads();                         // all waves done reading buf
  }

  // epilogue (C layout: VGPR r -> row half*8+r, col = l16 of 16x16 tile)
  #pragma unroll
  for (int t = 0; t < 4; ++t) {
    const int col = colBase + wcol * 64 + t * 16 + l16;
    const bool valid = col < NcolStore;
    const float bv = valid ? bias[col] : 0.0f;
    #pragma unroll
    for (int r = 0; r < 8; ++r) {
      const int row = rowBase + wrow * 16 + half * 8 + r;
      if (row >= MStore) continue;
      if constexpr (RELU_BF16_OUT) {
        if (col < ldc) {   // also zero-fill K-pad columns of hid
          float v = valid ? fmaxf(acc[t][r] + bv, 0.0f) : 0.0f;
          ((unsigned short*)Cout)[(size_t)row * ldc + col] = valid ? f2bf_bits(v) : 0;
        }
      } else {
        if (valid)
          ((float*)Cout)[(size_t)row * ldc + col] = acc[t][r] + bv;
      }
    }
  }
}

// ---------------------------------------------------------------- BatchNorm
__global__ __launch_bounds__(320) void bn_stats_kernel(
    const float* __restrict__ o, float* __restrict__ sum,
    float* __restrict__ sumsq, int N) {
  int c = threadIdx.x;
  if (c >= EMB) return;
  int r0 = blockIdx.x * 128;
  float s = 0.f, s2 = 0.f;
  for (int r = 0; r < 128; ++r) {
    int row = r0 + r;
    if (row < N) { float v = o[(size_t)row * EMB + c]; s += v; s2 += v * v; }
  }
  atomicAdd(&sum[c], s);
  atomicAdd(&sumsq[c], s2);
}

__global__ void bn_final_kernel(const float* __restrict__ sum,
                                const float* __restrict__ sumsq,
                                float* __restrict__ meanv,
                                float* __restrict__ rstdv, int N) {
  int c = blockIdx.x * blockDim.x + threadIdx.x;
  if (c >= EMB) return;
  float m = sum[c] / (float)N;
  float var = sumsq[c] / (float)N - m * m;
  meanv[c] = m;
  rstdv[c] = rsqrtf(var + BN_EPS);
}

__global__ void bn_apply_kernel(const float* __restrict__ o,
                                const float* __restrict__ meanv,
                                const float* __restrict__ rstdv,
                                const float* __restrict__ g,
                                const float* __restrict__ b,
                                float* __restrict__ dst, int N, int doRelu) {
  int i = blockIdx.x * blockDim.x + threadIdx.x;
  if (i >= N * EMB) return;
  int c = i % EMB;
  float v = (o[i] - meanv[c]) * rstdv[c] * g[c] + b[c];
  if (doRelu) v = fmaxf(v, 0.f);
  dst[i] = v;
}

// ---------------------------------------------------------------- launch
extern "C" void kernel_launch(void* const* d_in, const int* in_sizes, int n_in,
                              void* d_out, int out_size, void* d_ws, size_t ws_size,
                              hipStream_t stream) {
  (void)n_in; (void)out_size; (void)ws_size;
  const int*   x   = (const int*)d_in[0];
  const int*   ei  = (const int*)d_in[1];
  const int*   ea  = (const int*)d_in[2];
  const float* xe1 = (const float*)d_in[3];
  const float* xe2 = (const float*)d_in[4];
  const float* ee1 = (const float*)d_in[5];
  const float* ee2 = (const float*)d_in[6];
  const float* W1  = (const float*)d_in[7];
  const float* b1  = (const float*)d_in[8];
  const float* W2  = (const float*)d_in[9];
  const float* b2  = (const float*)d_in[10];
  const float* gam = (const float*)d_in[11];
  const float* bet = (const float*)d_in[12];

  const int N = in_sizes[0] / 2;
  const int E = in_sizes[1] / 2;
  const int NPAD = ((N + 63) / 64) * 64;

  // d_out [N,EMB] f32 doubles as the layer activation (h) buffer.
  float* hbuf = (float*)d_out;
  char* p = (char*)d_ws;
  size_t off = 0;
  auto carve = [&](size_t bytes) { char* q = p + off;
      off += (bytes + 255) & ~(size_t)255; return q; };
  float*          abuf   = (float*)         carve((size_t)N * EMB * 4);   // aggr / GEMM2 out
  unsigned short* aggrbf = (unsigned short*)carve((size_t)NPAD * KP1 * 2);
  unsigned short* hid    = (unsigned short*)carve((size_t)NPAD * KP2 * 2);
  unsigned short* Wp1    = (unsigned short*)carve((size_t)NCP1 * KP1 * 2);
  unsigned short* Wp2    = (unsigned short*)carve((size_t)NCP2 * KP2 * 2);
  float*          sum    = (float*)         carve(4 * EMB * 4);
  float* sumsq = sum + EMB;
  float* meanv = sum + 2 * EMB;
  float* rstdv = sum + 3 * EMB;

  const int NE = N * EMB;
  const int totalMsg = (E + N) * EMB;

  node_emb_kernel<<<(NE + 255) / 256, 256, 0, stream>>>(x, xe1, xe2, hbuf, N);

  for (int l = 0; l < NLAYER; ++l) {
    zero_kernel<<<(NE + 255) / 256, 256, 0, stream>>>(abuf, NE);
    zero_kernel<<<(2 * EMB + 255) / 256, 256, 0, stream>>>(sum, 2 * EMB);

    scatter_kernel<<<(totalMsg + 255) / 256, 256, 0, stream>>>(
        hbuf, ei, ea, ee1 + (size_t)l * 6 * EMB, ee2 + (size_t)l * 3 * EMB,
        abuf, N, E);

    cvt_a_kernel<<<(NPAD * KP1 + 255) / 256, 256, 0, stream>>>(abuf, aggrbf, N, NPAD);
    pack_w_kernel<<<(NCP1 * KP1 + 255) / 256, 256, 0, stream>>>(
        W1 + (size_t)l * EMB * HID, Wp1, EMB, HID, KP1, NCP1);
    pack_w_kernel<<<(NCP2 * KP2 + 255) / 256, 256, 0, stream>>>(
        W2 + (size_t)l * HID * EMB, Wp2, HID, EMB, KP2, NCP2);

    dim3 g1(NPAD / 64, NCP1 / 128);
    gemm_wmma_kernel<true><<<g1, 256, 0, stream>>>(
        aggrbf, Wp1, b1 + (size_t)l * HID, hid, KP1, KP2, HID, NPAD);

    dim3 g2(NPAD / 64, NCP2 / 128);
    gemm_wmma_kernel<false><<<g2, 256, 0, stream>>>(
        hid, Wp2, b2 + (size_t)l * EMB, abuf, KP2, EMB, EMB, N);

    bn_stats_kernel<<<(N + 127) / 128, 320, 0, stream>>>(abuf, sum, sumsq, N);
    bn_final_kernel<<<1, 320, 0, stream>>>(sum, sumsq, meanv, rstdv, N);
    bn_apply_kernel<<<(NE + 255) / 256, 256, 0, stream>>>(
        abuf, meanv, rstdv, gam + (size_t)l * EMB, bet + (size_t)l * EMB,
        hbuf, N, (l < NLAYER - 1) ? 1 : 0);
  }
}
